// LstmDecoder_1228360646869
// MI455X (gfx1250) — matrix operand order repaired and checked by reading, not compiled, not run on hardware
//
#include <hip/hip_runtime.h>

typedef __attribute__((ext_vector_type(16))) _Float16 v16h;
typedef __attribute__((ext_vector_type(8)))  float    v8f;

#define BATCH 32
#define TSEQ  160
#define DIN   512
#define HID   256
#define NCLS  6625
#define H3    768          // 3*HID
#define TB    (TSEQ*BATCH) // 5120 rows
#define MT_TB (TB/16)      // 320 row tiles
#define KC_D  (DIN/32)     // 16 K-chunks for D=512
#define KC_H  (HID/32)     // 8  K-chunks for H=256
#define KC_2H (2*HID/32)   // 16 K-chunks for 2H=512
#define NT_H3 (H3/16)      // 48 col tiles
#define NT_C  416          // col tiles for C, padded: 26 blocks * 16 tiles = 6656 >= 6625

// ---------------- fragment index helpers (CDNA5 ISA layouts) ----
// A matrix 16x32 f16: lane&15 = M row; per-lane half index i maps to
//   K = ((i>>3)<<4) | (i&7) | ((lane>>4)<<3)   (K 0..7,16..23 for lanes 0-15; +8 for 16-31)
__device__ __forceinline__ int akidx(int lane, int i) {
  return ((i >> 3) << 4) | (i & 7) | (((lane >> 4) & 1) << 3);
}
// B matrix 32x16 f16: lane&15 = N col; K same as A: 0..7,16..23 for lanes 0-15; +8 for 16-31
__device__ __forceinline__ int bkidx(int lane, int i) {
  return ((i >> 3) << 4) | (i & 7) | (((lane >> 4) & 1) << 3);
}

__device__ __forceinline__ v8f wmma16(v16h a, v16h b, v8f c) {
  return __builtin_amdgcn_wmma_f32_16x16x32_f16(false, a, false, b, (short)0, c,
                                                false, false);
}

// ---------------- swizzle kernels ------------------------------------------
// B-operand swizzle: src [K,N] f32 row-major -> dst [K/32][ntiles][32][16] f16
__global__ void k_swz_b(const float* __restrict__ src, _Float16* __restrict__ dst,
                        int K, int N, int ntiles) {
  int gid = blockIdx.x * blockDim.x + threadIdx.x;
  int total = (K >> 5) * ntiles * 32;
  if (gid >= total) return;
  int lane = gid & 31;
  int rest = gid >> 5;
  int nt = rest % ntiles;
  int kc = rest / ntiles;
  int n = nt * 16 + (lane & 15);
  _Float16* o = dst + (size_t)gid * 16;
  if (n < N) {
    const float* s = src + (size_t)(kc << 5) * N + n;
#pragma unroll
    for (int i = 0; i < 16; ++i)
      o[i] = (_Float16)s[(size_t)bkidx(lane, i) * N];
  } else {
#pragma unroll
    for (int i = 0; i < 16; ++i) o[i] = (_Float16)0.0f;
  }
}

// A-operand swizzle of x: x [B,T,D] f32 -> dst [320][16][32][16] f16, row r = t*32+b
__global__ void k_swz_x(const float* __restrict__ x, _Float16* __restrict__ dst) {
  int gid = blockIdx.x * blockDim.x + threadIdx.x;
  if (gid >= MT_TB * KC_D * 32) return;
  int lane = gid & 31;
  int kc = (gid >> 5) & (KC_D - 1);
  int mt = gid >> 9;
  int row = mt * 16 + (lane & 15);      // = t*32 + b
  int t = row >> 5, b = row & 31;
  const float* s = x + ((size_t)b * TSEQ + t) * DIN + (kc << 5);
  _Float16* o = dst + (size_t)gid * 16;
#pragma unroll
  for (int i = 0; i < 16; ++i) o[i] = (_Float16)s[akidx(lane, i)];
}

// A-operand swizzle of concat(h_f,h_b): hs* [T*B, 256] f32 -> dst [320][16][32][16]
__global__ void k_swz_out(const float* __restrict__ hsf, const float* __restrict__ hsb,
                          _Float16* __restrict__ dst) {
  int gid = blockIdx.x * blockDim.x + threadIdx.x;
  if (gid >= MT_TB * KC_2H * 32) return;
  int lane = gid & 31;
  int kc = (gid >> 5) & (KC_2H - 1);
  int mt = gid >> 9;
  int row = mt * 16 + (lane & 15);
  const float* s = (kc < KC_H ? hsf : hsb) + (size_t)row * HID + ((kc & (KC_H - 1)) << 5);
  _Float16* o = dst + (size_t)gid * 16;
#pragma unroll
  for (int i = 0; i < 16; ++i) o[i] = (_Float16)s[akidx(lane, i)];
}

// ---------------- GEMM 1: x_proj = x_t @ kernel + bias ----------------------
// grid (3, 80), block 256 (8 waves). Wave tile 32x64. reverse=1 remaps row
// tiles to time-reversed order (xp_b[t'] = x_t[T-1-t'] @ kernel_bwd).
__global__ __launch_bounds__(256)
void k_gemm_xproj(const _Float16* __restrict__ a_swz, const _Float16* __restrict__ b_swz,
                  const float* __restrict__ bias, float* __restrict__ xp, int reverse) {
  int lane = threadIdx.x & 31;
  int wave = threadIdx.x >> 5;
  int wm = wave >> 2, wn = wave & 3;
  int mt0 = blockIdx.y * 4 + wm * 2;
  int nt0 = blockIdx.x * 16 + wn * 4;

  float bv[4];
#pragma unroll
  for (int j = 0; j < 4; ++j) bv[j] = bias[(nt0 + j) * 16 + (lane & 15)];
  v8f acc[2][4];
#pragma unroll
  for (int i = 0; i < 2; ++i)
#pragma unroll
    for (int j = 0; j < 4; ++j)
#pragma unroll
      for (int e = 0; e < 8; ++e) acc[i][j][e] = bv[j];

  int mtA[2];
#pragma unroll
  for (int i = 0; i < 2; ++i) {
    int mt = mt0 + i;
    mtA[i] = reverse ? (318 - ((mt >> 1) << 1) + (mt & 1)) : mt;
  }

  for (int kc = 0; kc < KC_D; ++kc) {
    v16h a[2], b[4];
#pragma unroll
    for (int i = 0; i < 2; ++i)
      a[i] = *(const v16h*)(a_swz + (((size_t)mtA[i] * KC_D + kc) * 32 + lane) * 16);
#pragma unroll
    for (int j = 0; j < 4; ++j)
      b[j] = *(const v16h*)(b_swz + (((size_t)kc * NT_H3 + nt0 + j) * 32 + lane) * 16);
#pragma unroll
    for (int i = 0; i < 2; ++i)
#pragma unroll
      for (int j = 0; j < 4; ++j)
        acc[i][j] = wmma16(a[i], b[j], acc[i][j]);
  }

  int coff = lane & 15;
  int rsub = 8 * (lane >> 4);
#pragma unroll
  for (int i = 0; i < 2; ++i)
#pragma unroll
    for (int e = 0; e < 8; ++e) {
      int row = (mt0 + i) * 16 + e + rsub;
      float* orow = xp + (size_t)row * H3;
#pragma unroll
      for (int j = 0; j < 4; ++j) orow[(nt0 + j) * 16 + coff] = acc[i][j][e];
    }
}

// ---------------- GRU scan (both directions, grid.x = 2) --------------------
// 32 waves; wave (wm, nb): batch rows wm*16..+16, h columns nb*16..+16.
// h state lives in C-fragment registers; LDS holds h in A-fragment layout.
__global__ __launch_bounds__(1024)
void k_gru_scan(const _Float16* __restrict__ rkf, const _Float16* __restrict__ rkb,
                const float* __restrict__ rbf, const float* __restrict__ rbb,
                const float* __restrict__ xpf, const float* __restrict__ xpb,
                float* __restrict__ hsf, float* __restrict__ hsb) {
  int dir = blockIdx.x;
  const _Float16* rk = dir ? rkb : rkf;
  const float* rb = dir ? rbb : rbf;
  const float* xp = dir ? xpb : xpf;
  float* hs = dir ? hsb : hsf;

  __shared__ _Float16 hlds[2][KC_H][32][16];   // 16 KB: h in A-fragment layout

  int lane = threadIdx.x & 31;
  int wave = threadIdx.x >> 5;   // 0..31
  int wm = wave & 1;
  int nb = wave >> 1;            // 0..15

  for (int idx = threadIdx.x; idx < 2 * KC_H * 32 * 16; idx += 1024)
    ((_Float16*)hlds)[idx] = (_Float16)0.0f;

  int colh = nb * 16 + (lane & 15);          // h column of this lane
  int rsub = 8 * (lane >> 4);
  // LDS scatter constants: global h col cg = colh; kin = cg & 31
  int kc_w = nb >> 1;                         // destination K-chunk
  int ip   = ((nb & 1) << 3) | (lane & 7);    // half-index within v16h
  int lphi = ((lane >> 3) & 1) << 4;          // destination lane half

  float rbz = rb[colh], rbr = rb[HID + colh], rbh = rb[2 * HID + colh];

  v8f h;
#pragma unroll
  for (int e = 0; e < 8; ++e) h[e] = 0.0f;

  __syncthreads();

  for (int t = 0; t < TSEQ; ++t) {
    v8f az, ar, ah;
#pragma unroll
    for (int e = 0; e < 8; ++e) { az[e] = rbz; ar[e] = rbr; ah[e] = rbh; }

#pragma unroll
    for (int kc = 0; kc < KC_H; ++kc) {
      v16h a = *(const v16h*)&hlds[wm][kc][lane][0];
      const _Float16* bp = rk + (((size_t)kc * NT_H3) * 32 + lane) * 16;
      v16h bz = *(const v16h*)(bp + (size_t)(nb)      * 512);
      v16h br = *(const v16h*)(bp + (size_t)(nb + 16) * 512);
      v16h bh = *(const v16h*)(bp + (size_t)(nb + 32) * 512);
      az = wmma16(a, bz, az);
      ar = wmma16(a, br, ar);
      ah = wmma16(a, bh, ah);
    }
    __syncthreads();   // all waves done reading old h from LDS

    const float* xpt = xp + (size_t)t * BATCH * H3;
    int trow = dir ? (TSEQ - 1 - t) : t;     // bwd stores time-reversed (h_b[::-1])
    float* hbase = hs + (size_t)trow * BATCH * HID;
    v8f hn;
#pragma unroll
    for (int e = 0; e < 8; ++e) {
      int brow = wm * 16 + e + rsub;
      const float* xr = xpt + (size_t)brow * H3;
      float xz = xr[colh];
      float xrv = xr[HID + colh];
      float xh = xr[2 * HID + colh];
      float z = 1.0f / (1.0f + __expf(-(xz + az[e])));
      float r = 1.0f / (1.0f + __expf(-(xrv + ar[e])));
      float hh = tanhf(xh + r * ah[e]);
      float hv = z * h[e] + (1.0f - z) * hh;
      hn[e] = hv;
      hlds[wm][kc_w][lphi + e + rsub][ip] = (_Float16)hv;  // A-layout scatter
      hbase[(size_t)brow * HID + colh] = hv;               // stream to HBM
    }
    h = hn;
    __syncthreads();   // new h visible before next step's reads
  }
}

// ---------------- GEMM 3: logits = out @ W + b ------------------------------
// grid (26, 80), block 256. Wave tile 32x64, K = 512, N padded to 6656.
__global__ __launch_bounds__(256)
void k_gemm_logits(const _Float16* __restrict__ a_swz, const _Float16* __restrict__ w_swz,
                   const float* __restrict__ bias, float* __restrict__ out) {
  int lane = threadIdx.x & 31;
  int wave = threadIdx.x >> 5;
  int wm = wave >> 2, wn = wave & 3;
  int mt0 = blockIdx.y * 4 + wm * 2;
  int nt0 = blockIdx.x * 16 + wn * 4;

  v8f acc[2][4];
#pragma unroll
  for (int i = 0; i < 2; ++i)
#pragma unroll
    for (int j = 0; j < 4; ++j)
#pragma unroll
      for (int e = 0; e < 8; ++e) acc[i][j][e] = 0.0f;

  for (int kc = 0; kc < KC_2H; ++kc) {
    v16h a[2], b[4];
#pragma unroll
    for (int i = 0; i < 2; ++i)
      a[i] = *(const v16h*)(a_swz + (((size_t)(mt0 + i) * KC_2H + kc) * 32 + lane) * 16);
#pragma unroll
    for (int j = 0; j < 4; ++j)
      b[j] = *(const v16h*)(w_swz + (((size_t)kc * NT_C + nt0 + j) * 32 + lane) * 16);
#pragma unroll
    for (int i = 0; i < 2; ++i)
#pragma unroll
      for (int j = 0; j < 4; ++j)
        acc[i][j] = wmma16(a[i], b[j], acc[i][j]);
  }

  int coff = lane & 15;
  int rsub = 8 * (lane >> 4);
#pragma unroll
  for (int i = 0; i < 2; ++i)
#pragma unroll
    for (int e = 0; e < 8; ++e) {
      int row = (mt0 + i) * 16 + e + rsub;
      float* orow = out + (size_t)row * NCLS;
#pragma unroll
      for (int j = 0; j < 4; ++j) {
        int col = (nt0 + j) * 16 + coff;
        if (col < NCLS) orow[col] = acc[i][j][e] + bias[col];
      }
    }
}

// ---------------- host launcher --------------------------------------------
extern "C" void kernel_launch(void* const* d_in, const int* in_sizes, int n_in,
                              void* d_out, int out_size, void* d_ws, size_t ws_size,
                              hipStream_t stream) {
  const float* x   = (const float*)d_in[0];  // [B,T,D]
  const float* kf  = (const float*)d_in[1];  // [D,3H]
  const float* rkf = (const float*)d_in[2];  // [H,3H]
  const float* bf  = (const float*)d_in[3];  // [2,3H]
  const float* kb  = (const float*)d_in[4];
  const float* rkb = (const float*)d_in[5];
  const float* bb  = (const float*)d_in[6];
  const float* W   = (const float*)d_in[7];  // [2H,C]
  const float* bc  = (const float*)d_in[8];  // [C]
  float* out = (float*)d_out;                // [T*B, C]

  char* ws = (char*)d_ws;
  size_t off = 0;
  auto take = [&](size_t bytes) -> char* {
    char* p = ws + off;
    off += (bytes + 255) & ~(size_t)255;
    return p;
  };
  _Float16* x_swz   = (_Float16*)take((size_t)MT_TB * KC_D * 32 * 16 * 2);  // 5.24 MB
  _Float16* kf_swz  = (_Float16*)take((size_t)KC_D * NT_H3 * 32 * 16 * 2);  // 0.79 MB
  _Float16* kb_swz  = (_Float16*)take((size_t)KC_D * NT_H3 * 32 * 16 * 2);
  _Float16* rkf_swz = (_Float16*)take((size_t)KC_H * NT_H3 * 32 * 16 * 2);  // 0.39 MB
  _Float16* rkb_swz = (_Float16*)take((size_t)KC_H * NT_H3 * 32 * 16 * 2);
  _Float16* w_swz   = (_Float16*)take((size_t)KC_2H * NT_C * 32 * 16 * 2);  // 6.82 MB
  float*    xp_f    = (float*)take((size_t)TB * H3 * 4);                    // 15.7 MB
  float*    xp_b    = (float*)take((size_t)TB * H3 * 4);
  float*    hs_f    = (float*)take((size_t)TB * HID * 4);                   // 5.24 MB
  float*    hs_b    = (float*)take((size_t)TB * HID * 4);
  _Float16* out_swz = (_Float16*)take((size_t)MT_TB * KC_2H * 32 * 16 * 2); // 5.24 MB
  (void)in_sizes; (void)n_in; (void)out_size; (void)ws_size;

  dim3 blk(256);
  int tot_a = MT_TB * KC_D * 32;  // 163840 (same for KC_2H)
  k_swz_x<<<dim3((tot_a + 255) / 256), blk, 0, stream>>>(x, x_swz);

  auto swzb = [&](const float* src, _Float16* dst, int K, int N, int ntiles) {
    int tot = (K >> 5) * ntiles * 32;
    k_swz_b<<<dim3((tot + 255) / 256), blk, 0, stream>>>(src, dst, K, N, ntiles);
  };
  swzb(kf,  kf_swz,  DIN,     H3,   NT_H3);
  swzb(kb,  kb_swz,  DIN,     H3,   NT_H3);
  swzb(rkf, rkf_swz, HID,     H3,   NT_H3);
  swzb(rkb, rkb_swz, HID,     H3,   NT_H3);
  swzb(W,   w_swz,   2 * HID, NCLS, NT_C);

  // x-projection GEMMs (bias row 0 fused); bwd uses time-reversed row tiles
  k_gemm_xproj<<<dim3(3, 80), blk, 0, stream>>>(x_swz, kf_swz, bf, xp_f, 0);
  k_gemm_xproj<<<dim3(3, 80), blk, 0, stream>>>(x_swz, kb_swz, bb, xp_b, 1);

  // sequential recurrence, both directions concurrently (recurrent bias = row 1)
  k_gru_scan<<<dim3(2), dim3(1024), 0, stream>>>(rkf_swz, rkb_swz,
                                                 bf + H3, bb + H3,
                                                 xp_f, xp_b, hs_f, hs_b);

  // concat(h_f, h_b) -> A-fragment layout, then classifier GEMM
  k_swz_out<<<dim3((tot_a + 255) / 256), blk, 0, stream>>>(hs_f, hs_b, out_swz);
  k_gemm_logits<<<dim3(26, 80), blk, 0, stream>>>(out_swz, w_swz, bc, out);
}
